// EnhancedGATv2Model_2929167695953
// MI455X (gfx1250) — compile-verified, matrix operand order checked by reading, not run on hardware
//
#include <hip/hip_runtime.h>
#include <hip/hip_bf16.h>

// ---------------------------------------------------------------------------
// EnhancedGATv2 forward on MI455X (gfx1250), bf16 WMMA GEMMs + VALU epilogues.
// Tile staging uses double-buffered CDNA5 async global->LDS copies (ASYNCcnt),
// overlapping global fetch of tile t+1 with WMMA compute on tile t.
// ---------------------------------------------------------------------------

#define D_DIM 384
#define FFN_DIM 1536

typedef __attribute__((ext_vector_type(16))) __bf16 v16bf;
typedef __attribute__((ext_vector_type(8)))  float  v8f;

static __device__ __forceinline__ unsigned short f2bf(float f) {
    unsigned u = __float_as_uint(f);
    unsigned r = ((u >> 16) & 1u) + 0x7fffu;   // round-to-nearest-even
    return (unsigned short)((u + r) >> 16);
}

// Low 32 bits of a flat shared-aperture address are the LDS byte offset
// (CDNA5 ISA 10.2).
static __device__ __forceinline__ unsigned lds_off(const void* p) {
    return (unsigned)(size_t)p;
}

// async copy 16B global -> LDS, tracked by ASYNCcnt (no VGPR staging)
static __device__ __forceinline__ void async_cp16(unsigned lds, const void* gptr) {
    asm volatile("global_load_async_to_lds_b128 %0, %1, off"
                 :: "v"(lds), "v"(gptr) : "memory");
}

// ---------------------------------------------------------------------------
// Tiled bf16 GEMM:  C[M,Nn] = A[M,K](bf16,row-major) x Bt[Nn,K](bf16) + bias
// BM=128, BN=64, BK=64; 256 threads = 8 waves in a 4(M) x 2(N) grid,
// each wave computes a 32x32 patch = 2x2 v_wmma_f32_16x16x32_bf16 tiles,
// two WMMA K-steps per staged tile. Double-buffered async staging:
// per wave each stage issues exactly 6 async copies (4 A + 2 B), and async
// completions are in-order, so s_wait_asynccnt<=6 guarantees the current
// tile landed while the next tile's copies are still in flight.
// Requires K % 64 == 0 (all K here are 384 or 1536).
// ---------------------------------------------------------------------------
#define BM 128
#define BN 64
#define BK 64
#define LDA_S 72   // padded LDS row stride (bf16 elems) = 144B, 16B aligned
#define LDB_S 72

union Frag16 { v16bf v; uint4 u[2]; };

template<int ACT, bool OUT_F32, bool OUT_BF16>
__global__ __launch_bounds__(256)
void gemm_bf16_wmma(const unsigned short* __restrict__ A,
                    const unsigned short* __restrict__ Bt,
                    const float* __restrict__ bias,
                    float* __restrict__ Cf,
                    unsigned short* __restrict__ Cbf,
                    int M, int Nn, int K)
{
    __shared__ unsigned short As[2 * BM * LDA_S];
    __shared__ unsigned short Bs[2 * BN * LDB_S];

    const int tid   = threadIdx.x;
    const int lane  = tid & 31;
    const int wave  = tid >> 5;
    const int waveM = wave >> 1;      // 0..3
    const int waveN = wave & 1;       // 0..1
    const int m0 = blockIdx.y * BM;
    const int n0 = blockIdx.x * BN;

    v8f acc[2][2];
#pragma unroll
    for (int i = 0; i < 2; ++i)
#pragma unroll
        for (int j = 0; j < 2; ++j)
            acc[i][j] = (v8f){0.f,0.f,0.f,0.f,0.f,0.f,0.f,0.f};

    const int mrow = lane & 15;
    const int kb   = (lane < 16) ? 0 : 8;   // 16-bit A/B fragment K base

    // Stage one BKx{BM,BN} tile into LDS buffer `sel` (6 async copies/thread;
    // OOB rows are clamped: they only feed never-stored OOB output rows/cols).
    auto stage = [&](int k0, int sel) {
        unsigned short* Ab = &As[sel * (BM * LDA_S)];
        unsigned short* Bb = &Bs[sel * (BN * LDB_S)];
#pragma unroll
        for (int c = tid; c < BM * 8; c += 256) {
            int row = c >> 3, seg = c & 7;
            int gr = m0 + row; if (gr >= M) gr = M - 1;
            async_cp16(lds_off(&Ab[row * LDA_S + seg * 8]),
                       A + (size_t)gr * K + k0 + seg * 8);
        }
#pragma unroll
        for (int c = tid; c < BN * 8; c += 256) {
            int row = c >> 3, seg = c & 7;
            int gn = n0 + row; if (gn >= Nn) gn = Nn - 1;
            async_cp16(lds_off(&Bb[row * LDB_S + seg * 8]),
                       Bt + (size_t)gn * K + k0 + seg * 8);
        }
    };

    const int nTiles = K / BK;
    stage(0, 0);

    for (int t = 0; t < nTiles; ++t) {
        const int sel = t & 1;
        if (t + 1 < nTiles) {
            stage((t + 1) * BK, sel ^ 1);   // prefetch next tile (6 more asyncs)
            asm volatile("s_wait_asynccnt 0x6" ::: "memory");  // current landed
        } else {
            asm volatile("s_wait_asynccnt 0x0" ::: "memory");
        }
        __syncthreads();   // all waves' tile data visible

        const unsigned short* Ab = &As[sel * (BM * LDA_S)];
        const unsigned short* Bb = &Bs[sel * (BN * LDB_S)];

        // Two WMMA K-steps per staged tile.
#pragma unroll
        for (int kk = 0; kk < BK; kk += 32) {
            const int cb = kb + kk;   // per-lane K base within tile
            // Fragment: per lane, K = {cb..cb+7} and {cb+16..cb+23}
            Frag16 a[2], b[2];
#pragma unroll
            for (int i = 0; i < 2; ++i) {
                int r = waveM * 32 + i * 16 + mrow;
                const uint4* p = reinterpret_cast<const uint4*>(&Ab[r * LDA_S + cb]);
                a[i].u[0] = p[0];       // bytes +0  -> K cb..cb+7
                a[i].u[1] = p[2];       // bytes +32 -> K cb+16..cb+23
            }
#pragma unroll
            for (int i = 0; i < 2; ++i) {
                int r = waveN * 32 + i * 16 + mrow;
                const uint4* p = reinterpret_cast<const uint4*>(&Bb[r * LDB_S + cb]);
                b[i].u[0] = p[0];
                b[i].u[1] = p[2];
            }
#pragma unroll
            for (int mi = 0; mi < 2; ++mi)
#pragma unroll
                for (int ni = 0; ni < 2; ++ni)
                    acc[mi][ni] = __builtin_amdgcn_wmma_f32_16x16x32_bf16(
                        false, a[mi].v, false, b[ni].v,
                        (short)0, acc[mi][ni], false, false);
        }
        __syncthreads();   // done reading `sel`; next iter may overwrite it
    }

    // Epilogue: C/D layout -> lane%16 = col, vgpr r = row (lanes>=16 => +8)
    const int col16 = lane & 15;
#pragma unroll
    for (int mi = 0; mi < 2; ++mi)
#pragma unroll
        for (int ni = 0; ni < 2; ++ni) {
            int col = n0 + waveN * 32 + ni * 16 + col16;
            if (col >= Nn) continue;
            float bv = bias ? bias[col] : 0.f;
#pragma unroll
            for (int r = 0; r < 8; ++r) {
                int row = m0 + waveM * 32 + mi * 16 + ((lane < 16) ? r : 8 + r);
                if (row >= M) continue;
                float x = acc[mi][ni][r] + bv;
                if (ACT == 1) x = fmaxf(x, 0.f);
                if (OUT_F32)  Cf[(size_t)row * Nn + col] = x;
                if (OUT_BF16) Cbf[(size_t)row * Nn + col] = f2bf(x);
            }
        }
}

// ---------------------------------------------------------------------------
// Helpers
// ---------------------------------------------------------------------------
__global__ void cvt_bf16_kernel(const float* __restrict__ in,
                                unsigned short* __restrict__ out, size_t n) {
    size_t i = (size_t)blockIdx.x * blockDim.x + threadIdx.x;
    if (i < n) out[i] = f2bf(in[i]);
}

// W[k][n] (f32) -> Wt[n][k] (bf16)
__global__ void wprep_kernel(const float* __restrict__ W,
                             unsigned short* __restrict__ Wt, int K, int Nn) {
    size_t i = (size_t)blockIdx.x * blockDim.x + threadIdx.x;
    if (i < (size_t)K * Nn) {
        int k = (int)(i / Nn), n = (int)(i % Nn);
        Wt[(size_t)n * K + k] = f2bf(W[i]);
    }
}

__global__ void init_last_kernel(int* last, int n) {
    int i = blockIdx.x * blockDim.x + threadIdx.x;
    if (i < n) last[i] = -1;
}

__global__ void scatter_last_kernel(const int* __restrict__ dst, int* last, int e) {
    int i = blockIdx.x * blockDim.x + threadIdx.x;
    if (i < e) atomicMax(&last[dst[i]], i);
}

// Winning-edge attention + residual pre-sum:
//   hsum[n,:] = h[n,:] + res * h_upd[n,:]     (h_upd = attention or h itself)
// One thread per (node, head).
__global__ void edge_attn_kernel(const float* __restrict__ q,
                                 const float* __restrict__ k,
                                 const float* __restrict__ v,
                                 const float* __restrict__ hfc,
                                 const int* __restrict__ src,
                                 const int* __restrict__ last,
                                 const float* __restrict__ attn_bias,
                                 const float* __restrict__ attn_scale_p,
                                 const float* __restrict__ res_scale_p,
                                 float* __restrict__ hsum,
                                 int nNodes, int heads)
{
    int idx  = blockIdx.x * blockDim.x + threadIdx.x;
    int node = idx / heads;
    int hh   = idx - node * heads;
    if (node >= nNodes) return;
    const int hd = D_DIM / heads;
    const float res = res_scale_p[0];

    const float* hrow = hfc + (size_t)node * D_DIM + hh * hd;
    float*       orow = hsum + (size_t)node * D_DIM + hh * hd;
    int e = last[node];
    if (e < 0) {                              // no incoming edge: h_upd = h
        for (int d = 0; d < hd; ++d) orow[d] = hrow[d] * (1.f + res);
        return;
    }
    int s = src[e];
    const float scale = attn_scale_p[0];
    const float inv   = rsqrtf((float)hd);
    const float* qrow  = q + (size_t)node * D_DIM + hh * hd;
    const float* kbase = k + (size_t)s * D_DIM;
    const float* vbase = v + (size_t)s * D_DIM;

    float sc[8];
    float mx = -3.4e38f;
    for (int g = 0; g < heads; ++g) {
        float a = 0.f;
        const float* kr = kbase + g * hd;
        for (int d = 0; d < hd; ++d) a += qrow[d] * kr[d];
        a = a * inv * scale + attn_bias[hh * heads + g];
        sc[g] = a; mx = fmaxf(mx, a);
    }
    float se = 0.f;
    for (int g = 0; g < heads; ++g) { sc[g] = __expf(sc[g] - mx); se += sc[g]; }
    float rn = 1.f / se;
    for (int d = 0; d < hd; ++d) {
        float a = 0.f;
        for (int g = 0; g < heads; ++g) a += sc[g] * vbase[g * hd + d];
        orow[d] = hrow[d] + res * (a * rn);
    }
}

// LayerNorm over 384 with optional fused residual (A + rs*Badd) and ELU.
// One wave32 per row, 8 rows per 256-thread block.
__global__ __launch_bounds__(256)
void ln_kernel(const float* __restrict__ A, const float* __restrict__ Badd,
               const float* __restrict__ rs_p,
               const float* __restrict__ g, const float* __restrict__ b,
               float* __restrict__ outf, unsigned short* __restrict__ outbf,
               int rows, int act)
{
    int wave = threadIdx.x >> 5, lane = threadIdx.x & 31;
    int row = blockIdx.x * 8 + wave;
    if (row >= rows) return;
    const float rs = Badd ? rs_p[0] : 0.f;
    const float* ar = A + (size_t)row * D_DIM;
    const float* br = Badd ? Badd + (size_t)row * D_DIM : nullptr;

    float x[12];
    float s = 0.f;
#pragma unroll
    for (int i = 0; i < 12; ++i) {
        int c = lane + i * 32;
        float val = ar[c];
        if (br) val += rs * br[c];
        x[i] = val; s += val;
    }
#pragma unroll
    for (int o = 16; o >= 1; o >>= 1) s += __shfl_xor(s, o, 32);
    float mu = s * (1.f / D_DIM);
    float vs = 0.f;
#pragma unroll
    for (int i = 0; i < 12; ++i) { float d = x[i] - mu; vs += d * d; }
#pragma unroll
    for (int o = 16; o >= 1; o >>= 1) vs += __shfl_xor(vs, o, 32);
    float inv = rsqrtf(vs * (1.f / D_DIM) + 1e-5f);
#pragma unroll
    for (int i = 0; i < 12; ++i) {
        int c = lane + i * 32;
        float y = (x[i] - mu) * inv * g[c] + b[c];
        if (act == 1) y = (y > 0.f) ? y : (__expf(y) - 1.f);  // ELU
        if (outf)  outf[(size_t)row * D_DIM + c] = y;
        if (outbf) outbf[(size_t)row * D_DIM + c] = f2bf(y);
    }
}

// ---------------------------------------------------------------------------
// Host-side orchestration
// ---------------------------------------------------------------------------
static inline int cdiv(int a, int b) { return (a + b - 1) / b; }

extern "C" void kernel_launch(void* const* d_in, const int* in_sizes, int n_in,
                              void* d_out, int out_size, void* d_ws, size_t ws_size,
                              hipStream_t stream)
{
    // Input order: jax pytree flatten (sorted dict keys) of setup_inputs():
    //   0: dst[E], 1: feat[N*D],
    //   2..5: duration {b1,b2,w1,w2},
    //   6..62: layers[0..2] x 19 sorted keys,
    //   63..66: pitch {b1,b2,w1,w2}, 67..70: velocity {b1,b2,w1,w2}, 71: src[E]
    const int* dsti = (const int*)d_in[0];
    const float* feat = (const float*)d_in[1];
    const int* srci = (const int*)d_in[71];
    const int E = in_sizes[0];
    const int N = in_sizes[1] / D_DIM;

    const int LBASE = 6, LSTRIDE = 19;
    const int O_ATTN_BIAS = 0, O_ATTN_SCALE = 1, O_FC_B = 2, O_FC_W = 3,
              O_FFN_B1 = 4, O_FFN_B2 = 5, O_FFN_W1 = 6, O_FFN_W2 = 7,
              O_K_B = 8, O_K_W = 9, O_LN1_B = 10, O_LN1_G = 11,
              O_LN2_B = 12, O_LN2_G = 13, O_Q_B = 14, O_Q_W = 15,
              O_RES = 16, O_V_B = 17, O_V_W = 18;
    const int HEADS[3] = {8, 8, 1};

    float* dout = (float*)d_out;
    float* out_h   = dout;
    float* out_pit = dout + (size_t)N * D_DIM;
    float* out_dur = out_pit + (size_t)N * 88;
    float* out_vel = out_dur + (size_t)N * 32;

    // ---- workspace carve-up ----
    char* wp = (char*)d_ws;
    auto alloc = [&](size_t bytes) -> void* {
        void* p = (void*)wp;
        wp += (bytes + 255) & ~(size_t)255;
        return p;
    };
    size_t nd = (size_t)N * D_DIM;
    unsigned short* bfY = (unsigned short*)alloc(nd * 2);              // layer in / h out (bf16)
    unsigned short* bfW = (unsigned short*)alloc(nd * 2);              // fc out / decoder mid (bf16)
    unsigned short* bfM = (unsigned short*)alloc((size_t)N * FFN_DIM * 2); // ffn mid (bf16)
    float* hfc   = (float*)alloc(nd * 4);
    float* qb    = (float*)alloc(nd * 4);
    float* kb    = (float*)alloc(nd * 4);
    float* vb    = (float*)alloc(nd * 4);
    float* hsum  = (float*)alloc(nd * 4);
    float* h1    = (float*)alloc(nd * 4);
    float* ffno  = (float*)alloc(nd * 4);
    float* hout  = (float*)alloc(nd * 4);
    int*   last  = (int*)alloc((size_t)N * 4);
    unsigned short* wt_fc = (unsigned short*)alloc((size_t)D_DIM * D_DIM * 2);
    unsigned short* wt_q  = (unsigned short*)alloc((size_t)D_DIM * D_DIM * 2);
    unsigned short* wt_k  = (unsigned short*)alloc((size_t)D_DIM * D_DIM * 2);
    unsigned short* wt_v  = (unsigned short*)alloc((size_t)D_DIM * D_DIM * 2);
    unsigned short* wt_f1 = (unsigned short*)alloc((size_t)D_DIM * FFN_DIM * 2);
    unsigned short* wt_f2 = (unsigned short*)alloc((size_t)D_DIM * FFN_DIM * 2);
    unsigned short* wt_d2 = (unsigned short*)alloc((size_t)D_DIM * 88 * 2);
    (void)ws_size; (void)n_in; (void)out_size;

    auto wprep = [&](const float* W, unsigned short* Wt, int K, int Nn) {
        size_t n = (size_t)K * Nn;
        wprep_kernel<<<(unsigned)((n + 255) / 256), 256, 0, stream>>>(W, Wt, K, Nn);
    };
    auto gemm = [&](int act, bool of32, bool obf,
                    const unsigned short* A, const unsigned short* Bt,
                    const float* bias, float* Cf, unsigned short* Cbf,
                    int M, int Nn, int K) {
        dim3 grid(cdiv(Nn, BN), cdiv(M, BM));
        if (act == 0 && of32 && obf)
            gemm_bf16_wmma<0, true, true><<<grid, 256, 0, stream>>>(A, Bt, bias, Cf, Cbf, M, Nn, K);
        else if (act == 0 && of32 && !obf)
            gemm_bf16_wmma<0, true, false><<<grid, 256, 0, stream>>>(A, Bt, bias, Cf, Cbf, M, Nn, K);
        else /* act==1, bf16-only */
            gemm_bf16_wmma<1, false, true><<<grid, 256, 0, stream>>>(A, Bt, bias, Cf, Cbf, M, Nn, K);
    };

    // feat -> bf16
    cvt_bf16_kernel<<<(unsigned)((nd + 255) / 256), 256, 0, stream>>>(feat, bfY, nd);

    for (int L = 0; L < 3; ++L) {
        const int base = LBASE + L * LSTRIDE;
        auto P = [&](int o) { return (const float*)d_in[base + o]; };
        const int heads = HEADS[L];

        // weight prep (transpose + bf16)
        wprep(P(O_FC_W),  wt_fc, D_DIM, D_DIM);
        wprep(P(O_Q_W),   wt_q,  D_DIM, D_DIM);
        wprep(P(O_K_W),   wt_k,  D_DIM, D_DIM);
        wprep(P(O_V_W),   wt_v,  D_DIM, D_DIM);
        wprep(P(O_FFN_W1), wt_f1, D_DIM, FFN_DIM);
        wprep(P(O_FFN_W2), wt_f2, FFN_DIM, D_DIM);

        // h = x @ fc_w + fc_b     (f32 for residual path, bf16 for q/k/v GEMMs)
        gemm(0, true, true,  bfY, wt_fc, P(O_FC_B), hfc, bfW, N, D_DIM, D_DIM);
        // q/k/v
        gemm(0, true, false, bfW, wt_q, P(O_Q_B), qb, nullptr, N, D_DIM, D_DIM);
        gemm(0, true, false, bfW, wt_k, P(O_K_B), kb, nullptr, N, D_DIM, D_DIM);
        gemm(0, true, false, bfW, wt_v, P(O_V_B), vb, nullptr, N, D_DIM, D_DIM);

        // last-write-wins edge per dst node
        init_last_kernel<<<cdiv(N, 256), 256, 0, stream>>>(last, N);
        scatter_last_kernel<<<cdiv(E, 256), 256, 0, stream>>>(dsti, last, E);
        // attention on winning edges + residual pre-sum
        edge_attn_kernel<<<cdiv(N * heads, 256), 256, 0, stream>>>(
            qb, kb, vb, hfc, srci, last, P(O_ATTN_BIAS),
            P(O_ATTN_SCALE), P(O_RES), hsum, N, heads);

        // h1 = LN1(hsum)
        ln_kernel<<<cdiv(N, 8), 256, 0, stream>>>(
            hsum, nullptr, nullptr, P(O_LN1_G), P(O_LN1_B), h1, bfY, N, 0);

        // FFN: relu(h1@w1+b1) (bf16) @ w2 + b2 (f32)
        gemm(1, false, true, bfY, wt_f1, P(O_FFN_B1), nullptr, bfM, N, FFN_DIM, D_DIM);
        gemm(0, true, false, bfM, wt_f2, P(O_FFN_B2), ffno, nullptr, N, D_DIM, FFN_DIM);

        // h = LN2(h1 + res*ffn) [+ ELU between layers]; last layer -> d_out
        float* f32dst = (L < 2) ? hout : out_h;
        int act = (L < 2) ? 1 : 0;
        ln_kernel<<<cdiv(N, 8), 256, 0, stream>>>(
            h1, ffno, P(O_RES), P(O_LN2_G), P(O_LN2_B), f32dst, bfY, N, act);
    }

    // decoders: relu(h@w1+b1)@w2+b2
    struct Dec { int b1, b2, w1, w2, od; float* out; };
    Dec decs[3] = {
        {63, 64, 65, 66, 88, out_pit},   // pitch
        { 2,  3,  4,  5, 32, out_dur},   // duration
        {67, 68, 69, 70, 32, out_vel},   // velocity
    };
    for (int i = 0; i < 3; ++i) {
        const Dec& dd = decs[i];
        wprep((const float*)d_in[dd.w1], wt_fc, D_DIM, D_DIM);
        wprep((const float*)d_in[dd.w2], wt_d2, D_DIM, dd.od);
        gemm(1, false, true, bfY, wt_fc, (const float*)d_in[dd.b1],
             nullptr, bfW, N, D_DIM, D_DIM);
        gemm(0, true, false, bfW, wt_d2, (const float*)d_in[dd.b2],
             dd.out, nullptr, N, dd.od, D_DIM);
    }
}
